// LSTM_89910845375388
// MI455X (gfx1250) — compile-verified, matrix operand order checked by reading, not run on hardware
//
#include <hip/hip_runtime.h>
#include <hip/hip_bf16.h>
#include <stdint.h>

// ---------------------------------------------------------------------------
// Bidirectional 2-layer LSTM + linear head for MI455X (gfx1250).
//
//  * bf16 WMMA (v_wmma_f32_16x16x32_bf16) for all GEMMs, fp32 accumulate,
//    fp32 gate nonlinearities.
//  * x pre-converted/transposed to bf16 [S,B,IN] in a parallel pass so the
//    sequential recurrence loop is pure b128-load + WMMA (no cvt VALU).
//  * One persistent 512-thread workgroup per (layer, direction); h kept in
//    LDS (bf16, padded rows -> conflict-free ds_load_b128), c in VGPRs.
//  * Weights pre-packed once into per-lane WMMA B-fragment order so each
//    lane fetches 32 contiguous bytes per tile (global_load_b128 x2,
//    L2-resident: ~0.77 MB/step/dir vs 192 MB L2).
//  * Final linear = parallel WMMA GEMM, one block per timestep.
//
// Workspace (assumes ws_size >= ~160 MB): packed weights + biases + bf16 x
// + bf16 inter-layer activations.
// ---------------------------------------------------------------------------

typedef __attribute__((ext_vector_type(16))) __bf16 v16bf;
typedef __attribute__((ext_vector_type(8)))  float  f32x8;

union BFrag {
  v16bf v;
  unsigned short u[16];
  uint4 q[2];
};

__device__ inline unsigned short f2bf(float f) {
  union { float f; uint32_t u; } c; c.f = f;
  uint32_t u = c.u;
  uint32_t r = u + 0x7FFFu + ((u >> 16) & 1u);   // round-to-nearest-even
  return (unsigned short)(r >> 16);
}
__device__ inline float sigmoidf_(float x) { return 1.0f / (1.0f + __expf(-x)); }
__device__ inline float tanh_clamped(float x) {
  x = fminf(fmaxf(x, -15.0f), 15.0f);
  float e = __expf(2.0f * x);
  return (e - 1.0f) / (e + 1.0f);
}

// ---------------------------------------------------------------------------
// x: fp32 [B=64, S=1024, IN=128]  ->  bf16 [S, B, IN]
// One thread = 8 contiguous k elements (32B read, 16B write).
// ---------------------------------------------------------------------------
__global__ void pack_x_kernel(const float* __restrict__ x,
                              unsigned short* __restrict__ xbf) {
  constexpr int S = 1024, Bm = 64, IN = 128;
  int tid = blockIdx.x * blockDim.x + threadIdx.x;   // [0, S*Bm*IN/8)
  int k8 = tid & 15;           // IN/8 = 16
  int rest = tid >> 4;
  int b = rest & 63;
  int s = rest >> 6;
  if (s >= S) return;
  const float* src = x + ((size_t)b * S + s) * IN + k8 * 8;
  unsigned short* dst = xbf + ((size_t)s * Bm + b) * IN + k8 * 8;
#pragma unroll
  for (int e = 0; e < 8; ++e) dst[e] = f2bf(src[e]);
}

// ---------------------------------------------------------------------------
// Pack W^T into per-tile, per-lane WMMA B-fragment order (bf16).
// Element (d, kt, nt, lane, e) <- W[d][n][k] with
//   n = nt*16 + (lane&15)
//   k = kt*32 + (e<8 ? 8*(lane>>4)+e : 16 + 8*(lane>>4) + (e-8))
// k < KIN reads w_ih, else w_hh (H=256 columns).  whh may be null (linear).
// ---------------------------------------------------------------------------
__global__ void pack_w_kernel(const float* __restrict__ wih,
                              const float* __restrict__ whh,
                              unsigned short* __restrict__ wpack,
                              int KIN, int KT, int NT, int ndirs, int nrows) {
  int tid = blockIdx.x * blockDim.x + threadIdx.x;
  int total = ndirs * KT * NT * 32;
  if (tid >= total) return;
  int lane = tid & 31;
  int rest = tid >> 5;
  int nt = rest % NT; rest /= NT;
  int kt = rest % KT; rest /= KT;
  int d  = rest;
  int n    = nt * 16 + (lane & 15);
  int half = lane >> 4;
  unsigned short* dst = wpack + (size_t)tid * 16;
  const float* wih_d = wih + (size_t)d * nrows * KIN;
  const float* whh_d = whh ? whh + (size_t)d * nrows * 256 : nullptr;
#pragma unroll
  for (int e = 0; e < 16; ++e) {
    int k = kt * 32 + (e < 8 ? half * 8 + e : 16 + half * 8 + (e - 8));
    float w = (k < KIN) ? wih_d[(size_t)n * KIN + k]
                        : whh_d[(size_t)n * 256 + (k - KIN)];
    dst[e] = f2bf(w);
  }
}

__global__ void pack_bias_kernel(const float* __restrict__ bih,
                                 const float* __restrict__ bhh,
                                 float* __restrict__ bias, int n) {
  int i = blockIdx.x * blockDim.x + threadIdx.x;
  if (i < n) bias[i] = bih[i] + bhh[i];
}

// ---------------------------------------------------------------------------
// One persistent block per direction.  512 threads = 16 waves; wave j owns
// H-column tile j (16 of 256) for all 4 gates and all 4 M-tiles of B=64.
// Input activations: bf16 [S, B, KIN]   (KIN = 128 for layer0, 512 layer1).
// ---------------------------------------------------------------------------
template <int KIN>
__global__ __launch_bounds__(512)
void lstm_layer_kernel(const unsigned short* __restrict__ xin,
                       const float* __restrict__ h0,
                       const float* __restrict__ c0,
                       const unsigned short* __restrict__ wpack,
                       const float* __restrict__ bias,
                       unsigned short* __restrict__ out,
                       int layer) {
  constexpr int H = 256, Bm = 64, S = 1024;
  constexpr int KT  = (KIN + H) / 32;   // total K tiles
  constexpr int KTX = KIN / 32;         // K tiles covering the x part
  constexpr int HSTR = H + 8;           // +16B row pad: conflict-free b128
  __shared__ unsigned short h_buf[Bm * HSTR];

  const int d    = blockIdx.x;          // direction
  const int lane = threadIdx.x & 31;
  const int j    = threadIdx.x >> 5;    // wave id == H column tile
  const int half = lane >> 4;
  const int nn   = lane & 15;

  // h0 -> LDS (bf16)
  for (int i = threadIdx.x; i < Bm * H; i += blockDim.x) {
    int row = i >> 8, col = i & 255;
    h_buf[row * HSTR + col] =
        f2bf(h0[((size_t)(2 * layer + d) * Bm + row) * H + col]);
  }

  // c state in registers (C/D layout: element r -> M = mt*16 + r + 8*half)
  f32x8 creg[4];
#pragma unroll
  for (int mt = 0; mt < 4; ++mt)
#pragma unroll
    for (int r = 0; r < 8; ++r) {
      int row = mt * 16 + r + 8 * half;
      creg[mt][r] = c0[((size_t)(2 * layer + d) * Bm + row) * H + j * 16 + nn];
    }

  float bgate[4];
#pragma unroll
  for (int g = 0; g < 4; ++g)
    bgate[g] = bias[(size_t)d * 1024 + g * 256 + j * 16 + nn];

  __syncthreads();

  const unsigned short* wbase = wpack + (size_t)d * KT * 64 * 512;

  for (int t = 0; t < S; ++t) {
    const int s = (d == 0) ? t : (S - 1 - t);

    f32x8 acc[4][4];
#pragma unroll
    for (int g = 0; g < 4; ++g)
#pragma unroll
      for (int mt = 0; mt < 4; ++mt)
#pragma unroll
        for (int r = 0; r < 8; ++r) acc[g][mt][r] = bgate[g];

    const unsigned short* xrow = xin + (size_t)s * Bm * KIN;

    for (int kt = 0; kt < KT; ++kt) {
      // ---- load all fragments for this kt first (one big clause) ----
      BFrag a[4], b[4];
#pragma unroll
      for (int mt = 0; mt < 4; ++mt) {
        const int row = mt * 16 + nn;     // A: M = lane&15 (ISA 7.12.2)
        if (kt < KTX) {
          const int kb0 = kt * 32 + half * 8;
          const unsigned short* p = xrow + (size_t)row * KIN + kb0;
          a[mt].q[0] = *(const uint4*)(p);
          a[mt].q[1] = *(const uint4*)(p + 16);
        } else {
          const int kb0 = (kt - KTX) * 32 + half * 8;
          a[mt].q[0] = *(const uint4*)&h_buf[row * HSTR + kb0];
          a[mt].q[1] = *(const uint4*)&h_buf[row * HSTR + kb0 + 16];
        }
      }
#pragma unroll
      for (int g = 0; g < 4; ++g) {
        const unsigned short* bp =
            wbase + ((size_t)kt * 64 + (g * 16 + j)) * 512 + lane * 16;
        b[g].q[0] = *(const uint4*)(bp);
        b[g].q[1] = *(const uint4*)(bp + 8);
      }
      // ---- 16 WMMAs; B fragment g reused across the 4 M tiles ----
#pragma unroll
      for (int g = 0; g < 4; ++g)
#pragma unroll
        for (int mt = 0; mt < 4; ++mt)
          acc[g][mt] = __builtin_amdgcn_wmma_f32_16x16x32_bf16(
              false, a[mt].v, false, b[g].v, (short)0, acc[g][mt], false, false);
    }

    __syncthreads();  // all reads of h_buf for this step complete

    // i,f,g,o combine (fp32), update c, write h to LDS + global (bf16)
#pragma unroll
    for (int mt = 0; mt < 4; ++mt) {
#pragma unroll
      for (int r = 0; r < 8; ++r) {
        float iv = sigmoidf_(acc[0][mt][r]);
        float fv = sigmoidf_(acc[1][mt][r]);
        float gv = tanh_clamped(acc[2][mt][r]);
        float ov = sigmoidf_(acc[3][mt][r]);
        float cn = fv * creg[mt][r] + iv * gv;
        creg[mt][r] = cn;
        unsigned short hb = f2bf(ov * tanh_clamped(cn));
        int row = mt * 16 + r + 8 * half;
        h_buf[row * HSTR + j * 16 + nn] = hb;
        out[((size_t)s * Bm + row) * 512 + d * H + j * 16 + nn] = hb;
      }
    }
    __syncthreads();  // new h visible before next step's reads
  }
}

// ---------------------------------------------------------------------------
// Final linear: out[s,b,o] = l1_out[s,b,:512] @ lin_w^T + lin_b.
// One block per timestep; 4 waves; wave = M tile; N tiles 0..3; K tiles 16.
// ---------------------------------------------------------------------------
__global__ __launch_bounds__(128)
void linear_kernel(const unsigned short* __restrict__ xin,
                   const unsigned short* __restrict__ wpack,
                   const float* __restrict__ linb,
                   float* __restrict__ out) {
  constexpr int K = 512, KT = 16, Bm = 64;
  const int s    = blockIdx.x;
  const int lane = threadIdx.x & 31;
  const int mt   = threadIdx.x >> 5;
  const int half = lane >> 4;
  const int nn   = lane & 15;

  f32x8 acc[4];
#pragma unroll
  for (int nt = 0; nt < 4; ++nt) {
    float bv = linb[nt * 16 + nn];
#pragma unroll
    for (int r = 0; r < 8; ++r) acc[nt][r] = bv;
  }

  for (int kt = 0; kt < KT; ++kt) {
    BFrag a, b[4];
    const int row = mt * 16 + nn;
    const unsigned short* p = xin + ((size_t)s * Bm + row) * K + kt * 32 + half * 8;
    a.q[0] = *(const uint4*)(p);
    a.q[1] = *(const uint4*)(p + 16);
#pragma unroll
    for (int nt = 0; nt < 4; ++nt) {
      const unsigned short* bp = wpack + ((size_t)kt * 4 + nt) * 512 + lane * 16;
      b[nt].q[0] = *(const uint4*)(bp);
      b[nt].q[1] = *(const uint4*)(bp + 8);
    }
#pragma unroll
    for (int nt = 0; nt < 4; ++nt)
      acc[nt] = __builtin_amdgcn_wmma_f32_16x16x32_bf16(
          false, a.v, false, b[nt].v, (short)0, acc[nt], false, false);
  }
#pragma unroll
  for (int nt = 0; nt < 4; ++nt)
#pragma unroll
    for (int r = 0; r < 8; ++r) {
      int row = mt * 16 + r + 8 * half;
      out[((size_t)s * Bm + row) * 64 + nt * 16 + nn] = acc[nt][r];
    }
}

// ---------------------------------------------------------------------------
extern "C" void kernel_launch(void* const* d_in, const int* in_sizes, int n_in,
                              void* d_out, int out_size, void* d_ws, size_t ws_size,
                              hipStream_t stream) {
  (void)in_sizes; (void)n_in; (void)out_size; (void)ws_size;

  const float* x       = (const float*)d_in[0];   // [64,1024,128]
  const float* h0      = (const float*)d_in[1];   // [4,64,256]
  const float* c0      = (const float*)d_in[2];   // [4,64,256]
  const float* w_ih_l0 = (const float*)d_in[3];   // [2,1024,128]
  const float* w_hh_l0 = (const float*)d_in[4];   // [2,1024,256]
  const float* b_ih_l0 = (const float*)d_in[5];   // [2,1024]
  const float* b_hh_l0 = (const float*)d_in[6];
  const float* w_ih_l1 = (const float*)d_in[7];   // [2,1024,512]
  const float* w_hh_l1 = (const float*)d_in[8];   // [2,1024,256]
  const float* b_ih_l1 = (const float*)d_in[9];
  const float* b_hh_l1 = (const float*)d_in[10];
  const float* lin_w   = (const float*)d_in[11];  // [64,512]
  const float* lin_b   = (const float*)d_in[12];  // [64]

  // workspace carve-up (256B aligned)
  char* ws = (char*)d_ws;
  size_t off = 0;
  auto carve = [&](size_t bytes) -> char* {
    char* p = ws + off;
    off = (off + bytes + 255) & ~(size_t)255;
    return p;
  };
  unsigned short* wpack_l0  = (unsigned short*)carve((size_t)2 * 12 * 64 * 512 * 2);
  unsigned short* wpack_l1  = (unsigned short*)carve((size_t)2 * 24 * 64 * 512 * 2);
  unsigned short* wpack_lin = (unsigned short*)carve((size_t)16 * 4 * 512 * 2);
  float*          bias_l0   = (float*)carve(2048 * 4);
  float*          bias_l1   = (float*)carve(2048 * 4);
  unsigned short* xbf       = (unsigned short*)carve((size_t)1024 * 64 * 128 * 2);
  unsigned short* l0_out    = (unsigned short*)carve((size_t)1024 * 64 * 512 * 2);
  unsigned short* l1_out    = (unsigned short*)carve((size_t)1024 * 64 * 512 * 2);

  // parallel pre-passes: pack weights/biases, convert+transpose x
  {
    int totx = 1024 * 64 * 16;   // one thread per 8 elements
    pack_x_kernel<<<totx / 256, 256, 0, stream>>>(x, xbf);
    int tot0 = 2 * 12 * 64 * 32;
    pack_w_kernel<<<(tot0 + 255) / 256, 256, 0, stream>>>(
        w_ih_l0, w_hh_l0, wpack_l0, 128, 12, 64, 2, 1024);
    int tot1 = 2 * 24 * 64 * 32;
    pack_w_kernel<<<(tot1 + 255) / 256, 256, 0, stream>>>(
        w_ih_l1, w_hh_l1, wpack_l1, 512, 24, 64, 2, 1024);
    int totl = 1 * 16 * 4 * 32;
    pack_w_kernel<<<(totl + 255) / 256, 256, 0, stream>>>(
        lin_w, nullptr, wpack_lin, 512, 16, 4, 1, 64);
    pack_bias_kernel<<<8, 256, 0, stream>>>(b_ih_l0, b_hh_l0, bias_l0, 2048);
    pack_bias_kernel<<<8, 256, 0, stream>>>(b_ih_l1, b_hh_l1, bias_l1, 2048);
  }

  // layer 0 (both directions concurrently), then layer 1, then linear head
  lstm_layer_kernel<128><<<2, 512, 0, stream>>>(
      xbf, h0, c0, wpack_l0, bias_l0, l0_out, 0);
  lstm_layer_kernel<512><<<2, 512, 0, stream>>>(
      l0_out, h0, c0, wpack_l1, bias_l1, l1_out, 1);
  linear_kernel<<<1024, 128, 0, stream>>>(l1_out, wpack_lin, lin_b, (float*)d_out);
}
